// EVSSBlock_4312147165611
// MI455X (gfx1250) — compile-verified, matrix operand order checked by reading, not run on hardware
//
#include <hip/hip_runtime.h>
#include <cstdint>
#include <cstddef>

// EVSS / VMamba block for MI455X (gfx1250, wave32, WMMA bf16 + TDM).
#define B_   8
#define C_   192
#define P_   4096      // H*W, H=W=64
#define D_   192       // D_INNER
#define NST  4
#define RNK  12
#define KD   4
#define DI2  384       // 2*D_INNER

typedef __bf16 bf16_t;
typedef __attribute__((ext_vector_type(16))) __bf16 v16bf;
typedef __attribute__((ext_vector_type(8)))  __bf16 v8bf;
typedef __attribute__((ext_vector_type(8)))  float  v8f;
typedef __attribute__((ext_vector_type(4)))  unsigned int v4u;
typedef __attribute__((ext_vector_type(8)))  int v8i;
typedef __attribute__((ext_vector_type(4)))  int v4i;

__device__ __forceinline__ float sigmoidf_(float v){ return 1.0f/(1.0f+__expf(-v)); }

// direction k, sequence position l -> pixel index (H=W=64)
__device__ __forceinline__ int pix_of(int k, int l){
  int ll = (k >= 2) ? (P_-1-l) : l;
  if (k & 1) return ((ll & 63) << 6) | (ll >> 6);   // col-major traversal
  return ll;
}

// ---------- f32 -> bf16 weight conversion ----------
__global__ void k_cvt_bf16(const float* __restrict__ src, bf16_t* __restrict__ dst, int n){
  int i = blockIdx.x*blockDim.x + threadIdx.x;
  if (i < n) dst[i] = (bf16_t)src[i];
}

// ---------- LayerNorm over C; (B,C,P) f32 -> (B,P,C) bf16 via LDS transpose ----------
__global__ __launch_bounds__(64) void k_ln_in(const float* __restrict__ x,
    const float* __restrict__ lnw, const float* __restrict__ lnb,
    bf16_t* __restrict__ xn){
  __shared__ bf16_t sm[64*C_];
  int tile = blockIdx.x & 63;
  int b    = blockIdx.x >> 6;
  int p    = tile*64 + threadIdx.x;     // one pixel per thread, coalesced over p
  const float* xb = x + (size_t)b*C_*P_ + p;
  float s=0.f, q=0.f;
  for (int c=0;c<C_;++c){ float v = xb[(size_t)c*P_]; s+=v; q+=v*v; }
  float mu = s*(1.0f/C_);
  float rs = rsqrtf(q*(1.0f/C_) - mu*mu + 1e-5f);
  for (int c=0;c<C_;++c){
    float v = (xb[(size_t)c*P_]-mu)*rs*lnw[c]+lnb[c];
    sm[threadIdx.x*C_ + c] = (bf16_t)v;
  }
  __syncthreads();
  bf16_t* dst = xn + ((size_t)b*P_ + (size_t)tile*64)*C_;
  for (int i = threadIdx.x; i < 64*C_; i += 64) dst[i] = sm[i];
}

// ---------- bf16 WMMA GEMM with TDM-staged A slab ----------
// Out[M,N] = A[M,K] * W[N,K]^T, K == 192.
// Block = 8 waves = 128 rows of A. TDM loads the 128x192 bf16 slab to LDS once;
// each wave keeps its 6 A fragments in VGPRs and sweeps all N tiles.
__global__ __launch_bounds__(256) void k_gemm_bf16(
    const bf16_t* __restrict__ A, const bf16_t* __restrict__ Wt,
    float* __restrict__ Out, int N){
  extern __shared__ bf16_t smA[];             // 128*192 bf16 = 48 KB, LDS offset 0
  const int K = 192;
  int lane = threadIdx.x & 31;
  int wave = threadIdx.x >> 5;
  int m0blk = blockIdx.x << 7;                // 128 rows per block

  if (wave == 0) {                            // one wave drives the TDM
    unsigned long long ga =
        (unsigned long long)(uintptr_t)(A + (size_t)m0blk*K);
    v4u g0;
    g0[0] = 1u;                                   // count=1, gather off
    g0[1] = 0u;                                   // lds_addr = 0 (dynamic LDS base)
    g0[2] = (unsigned)(ga & 0xffffffffu);         // global_addr[31:0]
    g0[3] = (unsigned)((ga >> 32) & 0x1ffffffu)   // global_addr[56:32]
          | (2u << 30);                           // type=2 ("image")
    v8i g1;
    g1[0] = (2 << 16);                            // data_size=4B; mask/flags=0
    g1[1] = (96 << 16);                           // tensor_dim0[15:0]=96 dwords/row
    g1[2] = (128 << 16);                          // tensor_dim1[15:0]=128 rows
    g1[3] = (96 << 16);                           // tile_dim0=96
    g1[4] = 128;                                  // tile_dim1=128, tile_dim2=0
    g1[5] = 96;                                   // tensor_dim0_stride=96 (contiguous)
    g1[6] = 0;
    g1[7] = 0;
    v4i z4 = {0,0,0,0};
#if defined(__clang_major__) && (__clang_major__ >= 23)
    v8i z8 = {0,0,0,0,0,0,0,0};
    __builtin_amdgcn_tensor_load_to_lds(g0, g1, z4, z4, z8, 0);
#else
    __builtin_amdgcn_tensor_load_to_lds(g0, g1, z4, z4, 0);
#endif
    __builtin_amdgcn_s_wait_tensorcnt(0);
  }
  __syncthreads();

  int lm = lane & 15;
  int kh = (lane >> 4) << 3;                  // K-half select (0 or 8 bf16)
  // A fragments from LDS -> VGPRs, reused for every N tile
  v16bf afr[6];
  const bf16_t* arow = smA + (size_t)(wave*16 + lm)*K + kh;
  #pragma unroll
  for (int kc=0;kc<6;++kc){
    v8bf a0 = *(const v8bf*)(arow + kc*32);
    v8bf a1 = *(const v8bf*)(arow + kc*32 + 16);
    afr[kc] = __builtin_shufflevector(a0,a1,0,1,2,3,4,5,6,7,8,9,10,11,12,13,14,15);
  }

  int tilesN = N >> 4;
  for (int nt=0; nt<tilesN; ++nt){
    const bf16_t* bp = Wt + (size_t)(nt*16 + lm)*K + kh;
    __builtin_prefetch(bp, 0, 1);             // global_prefetch_b8 (L2 warm)
    v8f acc = {};
    #pragma unroll
    for (int kc=0;kc<6;++kc){
      v8bf b0 = *(const v8bf*)(bp + kc*32);
      v8bf b1 = *(const v8bf*)(bp + kc*32 + 16);
      v16bf bv = __builtin_shufflevector(b0,b1,0,1,2,3,4,5,6,7,8,9,10,11,12,13,14,15);
      acc = __builtin_amdgcn_wmma_f32_16x16x32_bf16(false, afr[kc], false, bv,
                                                    (short)0, acc, false, false);
    }
    float* op = Out + (size_t)(m0blk + wave*16 + ((lane>>4)<<3))*N + nt*16 + lm;
    #pragma unroll
    for (int r=0;r<8;++r) op[(size_t)r*N] = acc[r];
  }
}

// ---------- depthwise 3x3 + bias + SiLU: xz[:, :, 0:192] -> xcv (B,P,D) ----------
__global__ __launch_bounds__(192) void k_conv_silu(const float* __restrict__ xz,
    const float* __restrict__ cw, const float* __restrict__ cb,
    float* __restrict__ xcv){
  int d  = threadIdx.x;
  int bp = blockIdx.x;
  int p  = bp & 4095;
  int b  = bp >> 12;
  int h = p >> 6, w = p & 63;
  float acc = cb[d];
  #pragma unroll
  for (int i=0;i<3;++i){
    int hh = h + i - 1;
    if ((unsigned)hh >= 64u) continue;
    #pragma unroll
    for (int j=0;j<3;++j){
      int ww = w + j - 1;
      if ((unsigned)ww >= 64u) continue;
      acc += cw[d*9 + i*3 + j] * xz[((size_t)(b*P_ + ((hh<<6)|ww)))*DI2 + d];
    }
  }
  acc = acc * sigmoidf_(acc);
  xcv[((size_t)bp)*D_ + d] = acc;
}

// ---------- x_proj: xdbl[b,k,l,0:20] = x_proj_w[k] @ u(b,k,l,:) ----------
__global__ __launch_bounds__(320) void k_xdbl(const float* __restrict__ xcv,
    const float* __restrict__ xpw, float* __restrict__ xdbl){
  int t  = threadIdx.x;
  int li = t / 20, ci = t % 20;
  int idx   = blockIdx.x;                // (b*K + k)*256 + ltile
  int ltile = idx & 255;
  int bk    = idx >> 8;
  int kk = bk & 3, b = bk >> 2;
  int l = ltile*16 + li;
  int p = pix_of(kk, l);
  const float* up = xcv + ((size_t)(b*P_ + p))*D_;
  const float* wp = xpw + ((size_t)(kk*20 + ci))*D_;
  float acc = 0.f;
  for (int d0=0; d0<D_; ++d0) acc += up[d0]*wp[d0];
  xdbl[(((size_t)bk)*P_ + l)*20 + ci] = acc;
}

// ---------- selective scan: 1 thread per (b,k,d), sequential over L ----------
__global__ __launch_bounds__(192) void k_scan(const float* __restrict__ xdbl,
    const float* __restrict__ xcv,
    const float* __restrict__ dtw_g, const float* __restrict__ dtb_g,
    const float* __restrict__ alog, const float* __restrict__ ds_g,
    float* __restrict__ ys){
  int d  = threadIdx.x;
  int bk = blockIdx.x;          // b*K + k
  int kk = bk & 3, b = bk >> 2;
  int kd = kk*D_ + d;
  float dtw[RNK];
  #pragma unroll
  for (int r=0;r<RNK;++r) dtw[r] = dtw_g[kd*RNK + r];
  float dtb = dtb_g[kd];
  float An[NST];
  #pragma unroll
  for (int n=0;n<NST;++n) An[n] = -__expf(alog[kd*NST + n]);
  float Dsv = ds_g[kd];
  float h0=0.f,h1=0.f,h2=0.f,h3=0.f;
  const float* rowb = xdbl + ((size_t)bk)*P_*20;
  float* yb = ys + ((size_t)bk)*P_*D_ + d;
  for (int l=0;l<P_;++l){
    const float* row = rowb + l*20;
    float dtr = dtb;
    #pragma unroll
    for (int r=0;r<RNK;++r) dtr += dtw[r]*row[r];          // dt recomputed in-reg
    float dtv = (dtr > 20.f) ? dtr : log1pf(__expf(dtr));  // softplus
    int p = pix_of(kk, l);
    float u = xcv[((size_t)(b*P_ + p))*D_ + d];            // coalesced over d
    float y  = Dsv * u;
    float du = dtv * u;
    { float dA=__expf(dtv*An[0]); h0 = dA*h0 + du*row[12+0]; y += h0*row[16+0]; }
    { float dA=__expf(dtv*An[1]); h1 = dA*h1 + du*row[12+1]; y += h1*row[16+1]; }
    { float dA=__expf(dtv*An[2]); h2 = dA*h2 + du*row[12+2]; y += h2*row[16+2]; }
    { float dA=__expf(dtv*An[3]); h3 = dA*h3 + du*row[12+3]; y += h3*row[16+3]; }
    yb[(size_t)l*D_] = y;
  }
}

// ---------- cross-merge + LayerNorm(D) + SiLU gate -> bf16 (B,P,D) ----------
__global__ __launch_bounds__(256) void k_merge(const float* __restrict__ ys,
    const float* __restrict__ xz, const float* __restrict__ onw,
    const float* __restrict__ onb, bf16_t* __restrict__ yg){
  int lane = threadIdx.x & 31, wv = threadIdx.x >> 5;
  int pg = blockIdx.x*8 + wv;          // b*P + p ; one wave per pixel
  int b = pg >> 12, p = pg & 4095;
  int l1 = ((p & 63) << 6) | (p >> 6); // inverse col-major index (involution)
  size_t base = (size_t)(b*4)*P_*D_;
  float vals[6]; float s=0.f,q=0.f;
  #pragma unroll
  for (int j=0;j<6;++j){
    int d = lane + 32*j;
    float v = ys[base + ((size_t)0*P_ + p)*D_          + d]
            + ys[base + ((size_t)2*P_ + (P_-1-p))*D_   + d]
            + ys[base + ((size_t)1*P_ + l1)*D_         + d]
            + ys[base + ((size_t)3*P_ + (P_-1-l1))*D_  + d];
    vals[j]=v; s+=v; q+=v*v;
  }
  #pragma unroll
  for (int off=16; off>0; off>>=1){ s += __shfl_xor(s, off, 32); q += __shfl_xor(q, off, 32); }
  float mu = s*(1.0f/D_);
  float rs = rsqrtf(q*(1.0f/D_) - mu*mu + 1e-5f);
  const float* zp = xz + ((size_t)pg)*DI2 + D_;   // gate half of in_proj output
  bf16_t* op = yg + ((size_t)pg)*D_;
  #pragma unroll
  for (int j=0;j<6;++j){
    int d = lane + 32*j;
    float w  = (vals[j]-mu)*rs*onw[d] + onb[d];
    float zv = zp[d];
    op[d] = (bf16_t)(w * (zv * sigmoidf_(zv)));
  }
}

// ---------- residual + transpose back to (B,C,H,W) ----------
__global__ void k_residual(const float* __restrict__ x, const float* __restrict__ og,
    float* __restrict__ out){
  int i = blockIdx.x*blockDim.x + threadIdx.x;   // over B*C*P
  int p  = i & 4095;
  int bc = i >> 12;
  int c  = bc % C_;
  int b  = bc / C_;
  out[i] = x[i] + og[((size_t)(b*P_ + p))*C_ + c];
}

extern "C" void kernel_launch(void* const* d_in, const int* in_sizes, int n_in,
                              void* d_out, int out_size, void* d_ws, size_t ws_size,
                              hipStream_t stream){
  const float* x      = (const float*)d_in[0];
  const float* ln_w   = (const float*)d_in[1];
  const float* ln_b   = (const float*)d_in[2];
  const float* inpw   = (const float*)d_in[3];
  const float* convw  = (const float*)d_in[4];
  const float* convb  = (const float*)d_in[5];
  const float* xprojw = (const float*)d_in[6];
  const float* dtpw   = (const float*)d_in[7];
  const float* dtpb   = (const float*)d_in[8];
  const float* alog   = (const float*)d_in[9];
  const float* dsv    = (const float*)d_in[10];
  const float* onw    = (const float*)d_in[11];
  const float* onb    = (const float*)d_in[12];
  const float* outpw  = (const float*)d_in[13];
  float* out = (float*)d_out;

  char* ws = (char*)d_ws;
  size_t off = 0;
  auto carve = [&](size_t bytes)->char*{
    char* pp = ws + off; off = (off + bytes + 255) & ~(size_t)255; return pp;
  };
  bf16_t* wIn  = (bf16_t*)carve((size_t)DI2*C_*2);
  bf16_t* wOut = (bf16_t*)carve((size_t)C_*D_*2);
  bf16_t* xn   = (bf16_t*)carve((size_t)B_*P_*C_*2);
  float*  xz   = (float*) carve((size_t)B_*P_*DI2*4);
  float*  xcv  = (float*) carve((size_t)B_*P_*D_*4);
  float*  xdbl = (float*) carve((size_t)B_*KD*P_*20*4);
  float*  ysb  = (float*) carve((size_t)B_*KD*P_*D_*4);
  bf16_t* yg   = (bf16_t*)carve((size_t)B_*P_*D_*2);
  float*  og   = (float*) carve((size_t)B_*P_*C_*4);
  (void)ws_size; (void)in_sizes; (void)n_in; (void)out_size;

  const int slabLDS = 128*192*2;   // 48 KB dynamic LDS for the A slab

  k_cvt_bf16 <<<(DI2*C_+255)/256, 256, 0, stream>>>(inpw, wIn, DI2*C_);
  k_cvt_bf16 <<<(C_*D_ +255)/256, 256, 0, stream>>>(outpw, wOut, C_*D_);
  k_ln_in    <<<B_*64, 64, 0, stream>>>(x, ln_w, ln_b, xn);
  k_gemm_bf16<<<(B_*P_)/128, 256, slabLDS, stream>>>(xn, wIn, xz, DI2);
  k_conv_silu<<<B_*P_, D_, 0, stream>>>(xz, convw, convb, xcv);
  k_xdbl     <<<B_*KD*(P_/16), 320, 0, stream>>>(xcv, xprojw, xdbl);
  k_scan     <<<B_*KD, D_, 0, stream>>>(xdbl, xcv, dtpw, dtpb, alog, dsv, ysb);
  k_merge    <<<B_*P_/8, 256, 0, stream>>>(ysb, xz, onw, onb, yg);
  k_gemm_bf16<<<(B_*P_)/128, 256, slabLDS, stream>>>(yg, wOut, og, B_*P_ ? C_ : C_);
  k_residual <<<(B_*C_*P_)/256, 256, 0, stream>>>(x, og, out);
}
